// AttMultiHeadMultiResLoc_87832081203837
// MI455X (gfx1250) — matrix operand order repaired and checked by reading, not compiled, not run on hardware
//
#include <hip/hip_runtime.h>
#include <stdint.h>

// ---------------------------------------------------------------------------
// Types for CDNA5 WMMA
// ---------------------------------------------------------------------------
typedef __attribute__((ext_vector_type(16))) __bf16 v16bf;
typedef __attribute__((ext_vector_type(8)))  float  v8f;
typedef __attribute__((ext_vector_type(4)))  int    v4i;

#define B_    16
#define T_    2048
#define E_    1024
#define DK_   512
#define KAUG  1152           // 1024 enc dims + 128 conv-feature dims
#define M_    (B_ * T_)      // 32768 rows (b*t)

// ---------------------------------------------------------------------------
// Workspace layout (bytes)
// ---------------------------------------------------------------------------
#define OFF_ENC   ((size_t)0)                                  // bf16 enc       64 MB
#define OFF_A2    (OFF_ENC   + (size_t)M_ * E_ * 2)            // bf16 conv-feat 32 MB (x4 heads)
#define OFF_BMAT  (OFF_A2    + (size_t)4 * M_ * 128 * 2)       // bf16 [Wk|Watt] 4.5 MB
#define OFF_Q     (OFF_BMAT  + (size_t)4 * DK_ * KAUG * 2)     // f32 Q
#define OFF_PREF  (OFF_Q     + (size_t)4 * B_ * DK_ * 4)       // f32 filter prefix sums
#define OFF_EPART (OFF_PREF  + (size_t)4 * 128 * 256 * 4)      // f32 e partials (4 slots)
#define OFF_U     (OFF_EPART + (size_t)4 * M_ * 4 * 4)         // f32 u[h][b][e]
#define OFF_CCAT  (OFF_U     + (size_t)4 * B_ * E_ * 4)        // f32 c_cat[b][2048]

// ---------------------------------------------------------------------------
// Helpers
// ---------------------------------------------------------------------------
__device__ __forceinline__ unsigned short f2bf(float f) {
  unsigned u = __float_as_uint(f);
  unsigned r = (u + 0x7fffu + ((u >> 16) & 1u)) >> 16;   // RNE
  return (unsigned short)r;
}
__device__ __forceinline__ float bf2f(unsigned short h) {
  return __uint_as_float(((unsigned)h) << 16);
}

#if __has_builtin(__builtin_amdgcn_global_load_async_to_lds_b128)
#define HAVE_ASYNC 1
#else
#define HAVE_ASYNC 0
#endif

typedef __attribute__((address_space(1))) v4i gas_v4i;   // global int4
typedef __attribute__((address_space(3))) v4i las_v4i;   // LDS int4

__device__ __forceinline__ void cp16(void* ldst, const void* gsrc) {
#if HAVE_ASYNC
  __builtin_amdgcn_global_load_async_to_lds_b128(
      (gas_v4i*)gsrc, (las_v4i*)ldst, 0, 0);
#else
  *(uint4*)ldst = *(const uint4*)gsrc;
#endif
}

__device__ __forceinline__ void wait_async() {
#if HAVE_ASYNC
#if __has_builtin(__builtin_amdgcn_s_wait_asynccnt)
  __builtin_amdgcn_s_wait_asynccnt(0);
#else
  asm volatile("s_wait_asynccnt 0" ::: "memory");
#endif
#endif
}

// ---------------------------------------------------------------------------
// 1) fp32 -> bf16 cast of enc_hs_pad (vectorized, 4 elems/thread)
// ---------------------------------------------------------------------------
__global__ __launch_bounds__(256) void k_cast(const float* __restrict__ in,
                                              unsigned short* __restrict__ out,
                                              int n4) {
  int i = blockIdx.x * 256 + threadIdx.x;
  if (i >= n4) return;
  float4 v = ((const float4*)in)[i];
  ushort4 o;
  o.x = f2bf(v.x); o.y = f2bf(v.y); o.z = f2bf(v.z); o.w = f2bf(v.w);
  ((ushort4*)out)[i] = o;
}

// ---------------------------------------------------------------------------
// 2) Build augmented B matrix [Wk | Watt] per head, bf16, rows of length 1152
// ---------------------------------------------------------------------------
__global__ __launch_bounds__(256) void k_buildB(const float* __restrict__ Wk,
                                                const float* __restrict__ Watt,
                                                unsigned short* __restrict__ bm) {
  int idx = blockIdx.x * 256 + threadIdx.x;          // < 4*512*1152
  int h = idx / (DK_ * KAUG);
  int r = idx - h * DK_ * KAUG;
  int n = r / KAUG, c = r - n * KAUG;
  float v = (c < 1024) ? Wk[((size_t)(h * DK_ + n)) * E_ + c]
                       : Watt[((size_t)(h * DK_ + n)) * 128 + (c - 1024)];
  bm[idx] = f2bf(v);
}

// ---------------------------------------------------------------------------
// 3) Q[h,b,k] = dec_z . Wq + bq   (tiny; f32 VALU)
// ---------------------------------------------------------------------------
__global__ __launch_bounds__(256) void k_q(const float* __restrict__ dec_z,
                                           const float* __restrict__ Wq,
                                           const float* __restrict__ bq,
                                           float* __restrict__ Qd) {
  int idx = blockIdx.x * 256 + threadIdx.x;          // < 32768
  int h = idx >> 13;
  int r = idx & 8191;
  int b = r >> 9, k = r & 511;
  const float* wrow = Wq + ((size_t)(h * DK_ + k)) * E_;
  const float* z = dec_z + (size_t)b * E_;
  float acc = bq[h * DK_ + k];
  for (int d = 0; d < E_; ++d) acc += z[d] * wrow[d];
  Qd[idx] = acc;                                      // layout (h*16+b)*512+k
}

// ---------------------------------------------------------------------------
// 4) Prefix sums of conv filters (att_prev is a step fn => conv is analytic)
// ---------------------------------------------------------------------------
__global__ void k_prefix(const float* __restrict__ c0, const float* __restrict__ c1,
                         const float* __restrict__ c2, const float* __restrict__ c3,
                         float* __restrict__ pref) {
  int idx = blockIdx.x * blockDim.x + threadIdx.x;
  if (idx >= 512) return;
  int h = idx >> 7, c = idx & 127;
  const float* wsrc = (h == 0) ? c0 : (h == 1) ? c1 : (h == 2) ? c2 : c3;
  int af = 25 * (h + 1), L = 2 * af + 1;              // L = 51,101,151,201
  float* P = pref + (size_t)idx * 256;
  float run = 0.f;
  P[0] = 0.f;
  for (int i = 0; i < L; ++i) { run += wsrc[c * L + i]; P[i + 1] = run; }
}

// ---------------------------------------------------------------------------
// 5) Conv features a2[h][m][c] = (P[hi]-P[lo])/len, cast to bf16
// ---------------------------------------------------------------------------
__global__ __launch_bounds__(256) void k_a2(const float* __restrict__ pref,
                                            const int* __restrict__ lens,
                                            unsigned short* __restrict__ a2) {
  int idx = blockIdx.x * 256 + threadIdx.x;          // < 16777216
  int h = idx >> 22;
  int r = idx & 4194303;
  int m = r >> 7, c = r & 127;
  int b = m >> 11, t = m & 2047;
  int len = lens[b];
  int af = 25 * (h + 1), L = 2 * af + 1;
  int lo = af - t;       lo = lo < 0 ? 0 : lo;
  int hi = af + len - t; hi = hi > L ? L : hi;
  float v = 0.f;
  if (hi > lo) {
    const float* P = pref + ((size_t)(h * 128 + c)) * 256;
    v = (P[hi] - P[lo]) / (float)len;
  }
  a2[idx] = f2bf(v);
}

// ---------------------------------------------------------------------------
// 6) Main fused GEMM + score epilogue.
//    Per head: [32768 x 1152] x [1152 x 512] with bf16 WMMA, f32 accum.
//    Block: 128(M) x 128(N) tile, 8 waves (one 16-row m-tile each, all 128 n).
//    Double-buffered LDS staging via async global->LDS b128 copies;
//    B fragments software-pipelined one tile ahead to hide LDS latency.
//    Epilogue: e_part = sum_n tanh(S + Q[n]) * g[n]  (per 128-col slice).
// ---------------------------------------------------------------------------
#define ROWE 40   // padded LDS row (elements): 80 B, keeps 16B alignment

union Frag { uint4 q[2]; v16bf v; };

__global__ __launch_bounds__(256) void k_gemm_score(
    const unsigned short* __restrict__ encb,   // [32768][1024] bf16
    const unsigned short* __restrict__ a2,     // [4][32768][128] bf16
    const unsigned short* __restrict__ bmat,   // [4][512][1152] bf16
    const float* __restrict__ Qd,              // [(h*16+b)*512+k]
    const float* __restrict__ gw,              // [4][512]
    float* __restrict__ epart)                 // [4][32768][4]
{
  __shared__ __align__(16) unsigned short sA[2][128 * ROWE];
  __shared__ __align__(16) unsigned short sB[2][128 * ROWE];

  const int tid  = threadIdx.x;
  const int w    = tid >> 5;
  const int lane = tid & 31;
  const int l16  = lane & 15;
  const int hi   = lane >> 4;
  const int m0   = blockIdx.x * 128;
  const int nb   = blockIdx.y * 128;
  const int h    = blockIdx.z;
  const int bidx = m0 >> 11;                   // batch index (constant per block)

  auto stage = [&](int ks, int buf) {
    const int kk = ks * 32;
    for (int s = tid; s < 1024; s += 256) {
      if (s < 512) {                            // A tile: 128 rows x 32 cols
        int row = s >> 2, part = s & 3;
        const unsigned short* src;
        if (kk < 1024)
          src = encb + (size_t)(m0 + row) * E_ + kk + part * 8;
        else
          src = a2 + ((size_t)h * M_ + m0 + row) * 128 + (kk - 1024) + part * 8;
        cp16(&sA[buf][row * ROWE + part * 8], src);
      } else {                                  // B tile: 128 n-rows x 32 cols
        int bs = s - 512, row = bs >> 2, part = bs & 3;
        const unsigned short* src =
            bmat + ((size_t)(h * DK_ + nb + row)) * KAUG + kk + part * 8;
        cp16(&sB[buf][row * ROWE + part * 8], src);
      }
    }
  };

  v8f acc[8];
#pragma unroll
  for (int j = 0; j < 8; ++j)
#pragma unroll
    for (int e = 0; e < 8; ++e) acc[j][e] = 0.f;

  stage(0, 0);
  for (int ks = 0; ks < 36; ++ks) {
    wait_async();
    __syncthreads();
    if (ks + 1 < 36) stage(ks + 1, (ks + 1) & 1);   // overlap next stage with compute

    const unsigned short* As = sA[ks & 1];
    const unsigned short* Bs = sB[ks & 1];

    // A fragment (16x32): lanes 0-15 hold K 0..7 / 16..23, lanes 16-31 K 8..15 / 24..31
    Frag fa;
    const int arow = w * 16 + l16;
    fa.q[0] = *(const uint4*)(As + arow * ROWE + hi * 8);
    fa.q[1] = *(const uint4*)(As + arow * ROWE + 16 + hi * 8);

    // B fragments (32x16): lanes 0-15 hold K 0..7 / 16..23, lanes 16-31 K 8..15 / 24..31.
    // Software-pipelined one tile ahead so ds_load latency overlaps the WMMA.
    Frag fb0, fb1;
    fb0.q[0] = *(const uint4*)(Bs + l16 * ROWE + hi * 8);
    fb0.q[1] = *(const uint4*)(Bs + l16 * ROWE + 16 + hi * 8);
#pragma unroll
    for (int j = 0; j < 8; ++j) {
      Frag& cur = (j & 1) ? fb1 : fb0;
      Frag& nxt = (j & 1) ? fb0 : fb1;
      if (j < 7) {
        const int brow = (j + 1) * 16 + l16;
        nxt.q[0] = *(const uint4*)(Bs + brow * ROWE + hi * 8);
        nxt.q[1] = *(const uint4*)(Bs + brow * ROWE + 16 + hi * 8);
      }
      acc[j] = __builtin_amdgcn_wmma_f32_16x16x32_bf16(
          false, fa.v, false, cur.v, (short)0, acc[j], false, false);
    }
    __syncthreads();
  }

  // Epilogue: s[r] = sum over this wave's 128 n of tanh(S + Q[n]) * g[n]
  float s[8];
#pragma unroll
  for (int r = 0; r < 8; ++r) s[r] = 0.f;
#pragma unroll
  for (int j = 0; j < 8; ++j) {
    const int n = nb + j * 16 + l16;
    const float q = Qd[((size_t)(h * B_ + bidx)) * DK_ + n];
    const float g = gw[h * DK_ + n];
#pragma unroll
    for (int r = 0; r < 8; ++r) s[r] += tanhf(acc[j][r] + q) * g;
  }
#pragma unroll
  for (int r = 0; r < 8; ++r) {           // reduce across the 16 lanes of each half
    float v = s[r];
    v += __shfl_xor(v, 1, 32);
    v += __shfl_xor(v, 2, 32);
    v += __shfl_xor(v, 4, 32);
    v += __shfl_xor(v, 8, 32);
    s[r] = v;
  }
  if (l16 == 0) {
    const int mrow = m0 + w * 16 + hi * 8;  // lane0 -> rows r, lane16 -> rows r+8
#pragma unroll
    for (int r = 0; r < 8; ++r)
      epart[((size_t)h * M_ + mrow + r) * 4 + blockIdx.y] = s[r];
  }
}

// ---------------------------------------------------------------------------
// 7) Masked softmax over T per (h,b); writes attention weights to d_out
// ---------------------------------------------------------------------------
__global__ __launch_bounds__(256) void k_softmax(const float* __restrict__ epart,
                                                 const int* __restrict__ lens,
                                                 float* __restrict__ wout) {
  __shared__ float red[256];
  const int hb = blockIdx.x;                 // h*16 + b
  const int b = hb & 15;
  const int len = lens[b];
  const int tid = threadIdx.x;
  const float scaling = 0.04419417382415922f;   // 1/sqrt(512)

  float loc[8];
  float mymax = -3.4e38f;
#pragma unroll
  for (int i = 0; i < 8; ++i) {
    int t = tid + i * 256;
    const float* ep = epart + ((size_t)hb * T_ + t) * 4;
    float e = ep[0] + ep[1] + ep[2] + ep[3];
    float x = (t < len) ? e * scaling : -3.4e38f;
    loc[i] = x;
    mymax = fmaxf(mymax, x);
  }
  red[tid] = mymax; __syncthreads();
  for (int o = 128; o > 0; o >>= 1) {
    if (tid < o) red[tid] = fmaxf(red[tid], red[tid + o]);
    __syncthreads();
  }
  float gmax = red[0]; __syncthreads();

  float mysum = 0.f;
#pragma unroll
  for (int i = 0; i < 8; ++i) {
    int t = tid + i * 256;
    float v = (t < len) ? __expf(loc[i] - gmax) : 0.f;
    loc[i] = v; mysum += v;
  }
  red[tid] = mysum; __syncthreads();
  for (int o = 128; o > 0; o >>= 1) {
    if (tid < o) red[tid] += red[tid + o];
    __syncthreads();
  }
  float inv = 1.f / red[0];
#pragma unroll
  for (int i = 0; i < 8; ++i) {
    int t = tid + i * 256;
    wout[(size_t)hb * T_ + t] = loc[i] * inv;
  }
}

// ---------------------------------------------------------------------------
// 8) u[h,b,e] = sum_t w[h,b,t] * enc[b,t,e]  (V-GEMM eliminated by linearity)
// ---------------------------------------------------------------------------
__global__ __launch_bounds__(1024) void k_u(const unsigned short* __restrict__ encb,
                                            const float* __restrict__ wout,
                                            float* __restrict__ u) {
  __shared__ float sw[4][T_];
  const int b = blockIdx.x, tid = threadIdx.x;
  for (int i = tid; i < 4 * T_; i += 1024) {
    int h = i >> 11, t = i & (T_ - 1);
    sw[h][t] = wout[((size_t)h * B_ + b) * T_ + t];
  }
  __syncthreads();
  float a0 = 0.f, a1 = 0.f, a2v = 0.f, a3 = 0.f;
  const unsigned short* ep = encb + (size_t)b * T_ * E_ + tid;
  for (int t = 0; t < T_; ++t) {
    float x = bf2f(ep[(size_t)t * E_]);
    a0 += x * sw[0][t]; a1 += x * sw[1][t]; a2v += x * sw[2][t]; a3 += x * sw[3][t];
  }
  u[((size_t)0 * B_ + b) * E_ + tid] = a0;
  u[((size_t)1 * B_ + b) * E_ + tid] = a1;
  u[((size_t)2 * B_ + b) * E_ + tid] = a2v;
  u[((size_t)3 * B_ + b) * E_ + tid] = a3;
}

// ---------------------------------------------------------------------------
// 9) c_cat[b, h*512+v] = Wv[h,v,:] . u[h,b,:]
// ---------------------------------------------------------------------------
__global__ __launch_bounds__(256) void k_ccat(const float* __restrict__ Wv,
                                              const float* __restrict__ u,
                                              float* __restrict__ ccat) {
  int idx = blockIdx.x * 256 + threadIdx.x;      // < 32768
  int b = idx >> 11, j = idx & 2047;
  int h = j >> 9, v = j & 511;
  const float* wr = Wv + ((size_t)(h * 512 + v)) * E_;
  const float* ur = u + ((size_t)(h * B_ + b)) * E_;
  float acc = 0.f;
  for (int e = 0; e < E_; ++e) acc += wr[e] * ur[e];
  ccat[idx] = acc;
}

// ---------------------------------------------------------------------------
// 10) c[b,p] = sum_j Wo[p,j] * c_cat[b,j]
// ---------------------------------------------------------------------------
__global__ __launch_bounds__(256) void k_out(const float* __restrict__ Wo,
                                             const float* __restrict__ ccat,
                                             float* __restrict__ outc) {
  int idx = blockIdx.x * 256 + threadIdx.x;      // < 16384
  int b = idx >> 10, p = idx & 1023;
  const float* wr = Wo + (size_t)p * 2048;
  const float* cr = ccat + (size_t)b * 2048;
  float acc = 0.f;
  for (int j = 0; j < 2048; ++j) acc += wr[j] * cr[j];
  outc[idx] = acc;
}

// ---------------------------------------------------------------------------
// Launch
// ---------------------------------------------------------------------------
extern "C" void kernel_launch(void* const* d_in, const int* in_sizes, int n_in,
                              void* d_out, int out_size, void* d_ws, size_t ws_size,
                              hipStream_t stream) {
  const float* enc   = (const float*)d_in[0];
  const int*   lens  = (const int*)d_in[1];
  const float* dec_z = (const float*)d_in[2];
  const float* Wq    = (const float*)d_in[3];
  const float* bq    = (const float*)d_in[4];
  const float* Wk    = (const float*)d_in[5];
  const float* Wv    = (const float*)d_in[6];
  const float* gw    = (const float*)d_in[7];
  // d_in[8] = g_b: a per-head constant shift of e -> softmax-invariant, unused.
  const float* Watt  = (const float*)d_in[9];
  const float* cw0   = (const float*)d_in[10];
  const float* cw1   = (const float*)d_in[11];
  const float* cw2   = (const float*)d_in[12];
  const float* cw3   = (const float*)d_in[13];
  const float* Wo    = (const float*)d_in[14];

  char* ws = (char*)d_ws;
  unsigned short* encb  = (unsigned short*)(ws + OFF_ENC);
  unsigned short* a2    = (unsigned short*)(ws + OFF_A2);
  unsigned short* bmat  = (unsigned short*)(ws + OFF_BMAT);
  float* Qd    = (float*)(ws + OFF_Q);
  float* pref  = (float*)(ws + OFF_PREF);
  float* epart = (float*)(ws + OFF_EPART);
  float* u     = (float*)(ws + OFF_U);
  float* ccat  = (float*)(ws + OFF_CCAT);

  float* outc = (float*)d_out;                 // c: 16*1024
  float* wout = outc + B_ * E_;                // ws: 4*16*2048

  k_cast  <<<32768, 256, 0, stream>>>(enc, encb, (B_ * T_ * E_) / 4);
  k_buildB<<<(4 * DK_ * KAUG) / 256, 256, 0, stream>>>(Wk, Watt, bmat);
  k_q     <<<128, 256, 0, stream>>>(dec_z, Wq, bq, Qd);
  k_prefix<<<2, 256, 0, stream>>>(cw0, cw1, cw2, cw3, pref);
  k_a2    <<<(4 * M_ * 128) / 256, 256, 0, stream>>>(pref, lens, a2);

  dim3 gg(M_ / 128, 4, 4);                     // (m-blocks, n-blocks, heads)
  k_gemm_score<<<gg, 256, 0, stream>>>(encb, a2, bmat, Qd, gw, epart);

  k_softmax<<<64, 256, 0, stream>>>(epart, lens, wout);
  k_u      <<<B_, 1024, 0, stream>>>(encb, wout, u);
  k_ccat   <<<(B_ * 2048) / 256, 256, 0, stream>>>(Wv, u, ccat);
  k_out    <<<(B_ * E_) / 256, 256, 0, stream>>>(Wo, ccat, outc);
}
